// GraphQuantizer_50637664420102
// MI455X (gfx1250) — matrix operand order repaired and checked by reading, not compiled, not run on hardware
//
#include <hip/hip_runtime.h>
#include <math.h>

// ---------------- constants from the reference ----------------
#define NN      20480       // nodes
#define EEDGE   348160      // edges (N*DEG + N self loops)

typedef __attribute__((ext_vector_type(16))) _Float16 v16h;
typedef __attribute__((ext_vector_type(8)))  float    v8f;

// ---------------- helpers ----------------
__device__ __forceinline__ float warp_sum(float v) {
#pragma unroll
  for (int o = 16; o > 0; o >>= 1) v += __shfl_xor(v, o, 32);
  return v;
}

__device__ __forceinline__ void atomicMaxFloat(float* addr, float val) {
  int* ia = reinterpret_cast<int*>(addr);
  int old = __float_as_int(*addr);
  while (__int_as_float(old) < val) {
    int assumed = old;
    old = atomicCAS(ia, assumed, __float_as_int(val));
    if (old == assumed) break;
  }
}

// ---------------- generic fill ----------------
__global__ __launch_bounds__(256) void k_fill(float* p, float v, int n) {
  int i = blockIdx.x * blockDim.x + threadIdx.x;
  if (i < n) p[i] = v;
}

// ---------------- weight pack: f32 (KxN) -> f16 WMMA B-fragment order ----------------
// One thread per (ktile, ntile, lane): writes 16 contiguous halfs (32B) holding
// B[k0 + 8*hi + (i<8 ? i : i+8)][nt*16 + l] for i=0..15 (exact v_wmma_f32_16x16x32_f16 B layout).
__global__ __launch_bounds__(256) void k_packB(
    const float* __restrict__ W, _Float16* __restrict__ Wp, int K, int Nn)
{
  int total = (K >> 5) * (Nn >> 4) * 32;
  int idx = blockIdx.x * blockDim.x + threadIdx.x;
  if (idx >= total) return;
  int lane = idx & 31;
  int tile = idx >> 5;
  int ntiles = Nn >> 4;
  int nt = tile % ntiles;
  int kt = tile / ntiles;
  int hi = lane >> 4, l = lane & 15;
  _Float16* out = Wp + (size_t)idx * 16;
  const float* srcp = W + (size_t)(kt * 32 + hi * 8) * Nn + nt * 16 + l;
#pragma unroll
  for (int i = 0; i < 16; ++i) out[i] = (_Float16)srcp[(size_t)(i < 8 ? i : i + 8) * Nn];
}

// ---------------- WMMA GEMM: C[MxN] = A[MxK]@Bpacked + bias, opt GELU ----------------
// One wave per 16x64 output tile: shared A fragment, 4 accumulators, 4 WMMA/k-step.
// K multiple of 32; M multiple of 16; N multiple of 64.
__global__ __launch_bounds__(32) void k_gemm_bias(
    const float* __restrict__ A, const _Float16* __restrict__ Bp,
    const float* __restrict__ bias, float* __restrict__ Cm,
    int M, int K, int Nn, int act)
{
  const int tm   = blockIdx.x * 16;
  const int tn0  = blockIdx.y * 64;
  const int lane = threadIdx.x & 31;
  const int hi   = lane >> 4;
  const int l    = lane & 15;
  const int ntiles = Nn >> 4;

  v8f acc0 = {}, acc1 = {}, acc2 = {}, acc3 = {};
  for (int k0 = 0; k0 < K; k0 += 32) {
    const int kt = k0 >> 5;
    // A fragment: row M = tm + l; lane-half hi holds K runs [8hi,8hi+7] and [16+8hi,16+8hi+7]
    const float* arow = A + (size_t)(tm + l) * K + k0 + (hi << 3);
    float4 a0 = *(const float4*)(arow);
    float4 a1 = *(const float4*)(arow + 4);
    float4 a2 = *(const float4*)(arow + 16);
    float4 a3 = *(const float4*)(arow + 20);
    v16h af;
    af[0]  = (_Float16)a0.x; af[1]  = (_Float16)a0.y; af[2]  = (_Float16)a0.z; af[3]  = (_Float16)a0.w;
    af[4]  = (_Float16)a1.x; af[5]  = (_Float16)a1.y; af[6]  = (_Float16)a1.z; af[7]  = (_Float16)a1.w;
    af[8]  = (_Float16)a2.x; af[9]  = (_Float16)a2.y; af[10] = (_Float16)a2.z; af[11] = (_Float16)a2.w;
    af[12] = (_Float16)a3.x; af[13] = (_Float16)a3.y; af[14] = (_Float16)a3.z; af[15] = (_Float16)a3.w;

    const _Float16* bbase = Bp + ((size_t)kt * ntiles + (tn0 >> 4)) * 32 * 16 + (size_t)lane * 16;
    v16h b0 = *(const v16h*)(bbase);
    v16h b1 = *(const v16h*)(bbase + 32 * 16);
    v16h b2 = *(const v16h*)(bbase + 2 * 32 * 16);
    v16h b3 = *(const v16h*)(bbase + 3 * 32 * 16);
    acc0 = __builtin_amdgcn_wmma_f32_16x16x32_f16(false, af, false, b0, (short)0, acc0, false, false);
    acc1 = __builtin_amdgcn_wmma_f32_16x16x32_f16(false, af, false, b1, (short)0, acc1, false, false);
    acc2 = __builtin_amdgcn_wmma_f32_16x16x32_f16(false, af, false, b2, (short)0, acc2, false, false);
    acc3 = __builtin_amdgcn_wmma_f32_16x16x32_f16(false, af, false, b3, (short)0, acc3, false, false);
  }
#pragma unroll
  for (int t = 0; t < 4; ++t) {
    v8f acc = (t == 0) ? acc0 : (t == 1) ? acc1 : (t == 2) ? acc2 : acc3;
    int col = tn0 + t * 16 + l;
    float bv = bias ? bias[col] : 0.0f;
#pragma unroll
    for (int v = 0; v < 8; ++v) {
      int row = tm + v + (hi << 3);
      float x = acc[v] + bv;
      if (act == 1) x = 0.5f * x * (1.0f + erff(x * 0.70710678118654752f)); // exact GELU
      Cm[(size_t)row * Nn + col] = x;
    }
  }
}

// ---------------- up-projection (K=10 one-hot, tiny) ----------------
__global__ __launch_bounds__(256) void k_up(
    const float* __restrict__ feats, const float* __restrict__ Ws,
    const float* __restrict__ bs, const float* __restrict__ Wd,
    const float* __restrict__ bd, float* __restrict__ fs,
    float* __restrict__ fd, int n)
{
  int i = blockIdx.x * blockDim.x + threadIdx.x;
  if (i >= n) return;
  int node = i >> 6, c = i & 63;
  const float* f = feats + (size_t)node * 10;
  float s = bs[c], d = bd[c];
#pragma unroll
  for (int k = 0; k < 10; ++k) {
    float fv = f[k];
    s += fv * Ws[k * 64 + c];
    d += fv * Wd[k * 64 + c];
  }
  fs[i] = s; fd[i] = d;
}

// ---------------- GATv2 edge passes (one wave per edge) ----------------
__global__ __launch_bounds__(256) void k_edge_logits(
    const float* __restrict__ fs, const float* __restrict__ fd,
    const float* __restrict__ a, const int* __restrict__ src,
    const int* __restrict__ dst, float* __restrict__ logits,
    float* __restrict__ m, int nEdges, int Hd)
{
  int wid  = (blockIdx.x * blockDim.x + threadIdx.x) >> 5;
  int lane = threadIdx.x & 31;
  if (wid >= nEdges) return;
  int s = src[wid], d = dst[wid];
  int HC = Hd * 64;
  for (int h = 0; h < Hd; ++h) {
    const float* ps = fs + (size_t)s * HC + h * 64;
    const float* pd = fd + (size_t)d * HC + h * 64;
    const float* pa = a + h * 64;
    float acc = 0.f;
#pragma unroll
    for (int j = 0; j < 2; ++j) {
      int c = lane + 32 * j;
      float v = ps[c] + pd[c];
      v = v > 0.f ? v : 0.2f * v;          // leaky_relu slope 0.2
      acc += v * pa[c];
    }
    float tot = warp_sum(acc);
    if (lane == 0) {
      logits[(size_t)wid * Hd + h] = tot;
      atomicMaxFloat(&m[d * Hd + h], tot);
    }
  }
}

__global__ __launch_bounds__(256) void k_edge_exp(
    float* __restrict__ logits, const float* __restrict__ m,
    float* __restrict__ den, const int* __restrict__ dst, int nEdges, int Hd)
{
  int i = blockIdx.x * blockDim.x + threadIdx.x;
  if (i >= nEdges * Hd) return;
  int e = i / Hd, h = i - e * Hd;
  int d = dst[e];
  float ex = expf(logits[i] - m[d * Hd + h]);
  logits[i] = ex;
  atomicAdd(&den[d * Hd + h], ex);
}

__global__ __launch_bounds__(256) void k_edge_agg(
    const float* __restrict__ fs, const float* __restrict__ w,
    const int* __restrict__ src, const int* __restrict__ dst,
    float* __restrict__ out, int nEdges, int Hd)
{
  int wid  = (blockIdx.x * blockDim.x + threadIdx.x) >> 5;
  int lane = threadIdx.x & 31;
  if (wid >= nEdges) return;
  int s = src[wid], d = dst[wid];
  int HC = Hd * 64;
  for (int h = 0; h < Hd; ++h) {
    float ww = w[(size_t)wid * Hd + h];
#pragma unroll
    for (int j = 0; j < 2; ++j) {
      int c = lane + 32 * j;
      atomicAdd(&out[(size_t)d * HC + h * 64 + c],
                ww * fs[(size_t)s * HC + h * 64 + c]);
    }
  }
}

__global__ __launch_bounds__(256) void k_scale_rows(
    float* __restrict__ x, const float* __restrict__ den, int n, int Hd)
{
  int i = blockIdx.x * blockDim.x + threadIdx.x;
  if (i >= n) return;
  int HC = Hd * 64;
  int node = i / HC;
  int h = (i - node * HC) >> 6;
  x[i] /= den[node * Hd + h];
}

// ---------------- LayerNorm over last dim (64), one wave per row ----------------
__global__ __launch_bounds__(256) void k_layernorm(
    float* __restrict__ x, const float* __restrict__ g,
    const float* __restrict__ b, int rows)
{
  int wid  = (blockIdx.x * blockDim.x + threadIdx.x) >> 5;
  int lane = threadIdx.x & 31;
  if (wid >= rows) return;
  float* r = x + (size_t)wid * 64;
  float v0 = r[lane], v1 = r[lane + 32];
  float mu = warp_sum(v0 + v1) * (1.0f / 64.0f);
  float d0 = v0 - mu, d1 = v1 - mu;
  float var = warp_sum(d0 * d0 + d1 * d1) * (1.0f / 64.0f);
  float inv = rsqrtf(var + 1e-5f);
  r[lane]      = d0 * inv * g[lane]      + b[lane];
  r[lane + 32] = d1 * inv * g[lane + 32] + b[lane + 32];
}

// ---------------- global attention pooling ----------------
__global__ __launch_bounds__(256) void k_gate(
    const float* __restrict__ x, const float* __restrict__ Wg,
    const float* __restrict__ bg, const int* __restrict__ gid,
    float* __restrict__ g, float* __restrict__ gm, int nNodes)
{
  int wid  = (blockIdx.x * blockDim.x + threadIdx.x) >> 5;
  int lane = threadIdx.x & 31;
  if (wid >= nNodes) return;
  const float* r = x + (size_t)wid * 512;
  float acc = 0.f;
#pragma unroll
  for (int j = 0; j < 16; ++j) acc += r[lane + 32 * j] * Wg[lane + 32 * j];
  float t = warp_sum(acc);
  if (lane == 0) {
    t += bg[0];
    g[wid] = t;
    atomicMaxFloat(&gm[gid[wid]], t);
  }
}

__global__ __launch_bounds__(256) void k_gate_exp(
    float* __restrict__ g, const float* __restrict__ gm,
    float* __restrict__ gden, const int* __restrict__ gid, int nNodes)
{
  int i = blockIdx.x * blockDim.x + threadIdx.x;
  if (i >= nNodes) return;
  float ge = expf(g[i] - gm[gid[i]]);
  g[i] = ge;
  atomicAdd(&gden[gid[i]], ge);
}

__global__ __launch_bounds__(256) void k_pool(
    const float* __restrict__ x, const float* __restrict__ g,
    const float* __restrict__ gden, const int* __restrict__ gid,
    float* __restrict__ pooled, int nNodes)
{
  int i = blockIdx.x * blockDim.x + threadIdx.x;
  if (i >= nNodes * 512) return;
  int n = i >> 9, k = i & 511;
  int gi = gid[n];
  atomicAdd(&pooled[gi * 512 + k], (g[n] / gden[gi]) * x[i]);
}

__global__ __launch_bounds__(256) void k_tanh_out(
    const float* __restrict__ pooled, float* __restrict__ out, int n)
{
  int i = blockIdx.x * blockDim.x + threadIdx.x;
  if (i < n) out[i] = tanhf(pooled[i]);
}

// ---------------- host orchestration ----------------
extern "C" void kernel_launch(void* const* d_in, const int* in_sizes, int n_in,
                              void* d_out, int out_size, void* d_ws, size_t ws_size,
                              hipStream_t stream) {
  (void)n_in; (void)in_sizes; (void)ws_size;
  const int N = NN, Eed = EEDGE;

  const float* feats = (const float*)d_in[0];
  const int*   src   = (const int*)d_in[1];
  const int*   dst   = (const int*)d_in[2];
  const int*   gid   = (const int*)d_in[3];
  // params in setup_inputs() dict insertion order
  const float* upWs = (const float*)d_in[4];
  const float* upbs = (const float*)d_in[5];
  const float* upWd = (const float*)d_in[6];
  const float* upbd = (const float*)d_in[7];
  const float* upa  = (const float*)d_in[8];
  const float* a0Ws = (const float*)d_in[9];
  const float* a0bs = (const float*)d_in[10];
  const float* a0Wd = (const float*)d_in[11];
  const float* a0bd = (const float*)d_in[12];
  const float* a0a  = (const float*)d_in[13];
  const float* f0g  = (const float*)d_in[14];
  const float* f0b  = (const float*)d_in[15];
  const float* f0W1 = (const float*)d_in[16];
  const float* f0b1 = (const float*)d_in[17];
  const float* f0W2 = (const float*)d_in[18];
  const float* f0b2 = (const float*)d_in[19];
  const float* f0W3 = (const float*)d_in[20];
  const float* f0b3 = (const float*)d_in[21];
  const float* sWs  = (const float*)d_in[22];
  const float* sbs  = (const float*)d_in[23];
  const float* sWd  = (const float*)d_in[24];
  const float* sbd  = (const float*)d_in[25];
  const float* sa   = (const float*)d_in[26];
  const float* fsg  = (const float*)d_in[27];
  const float* fsb  = (const float*)d_in[28];
  const float* fsW1 = (const float*)d_in[29];
  const float* fsb1 = (const float*)d_in[30];
  const float* fsW2 = (const float*)d_in[31];
  const float* fsb2 = (const float*)d_in[32];
  const float* Wg   = (const float*)d_in[33];
  const float* bg   = (const float*)d_in[34];

  float* ws = (float*)d_ws;
  size_t off = 0;
  auto alloc = [&](size_t n) {                 // 32B-aligned float allocator
    off = (off + 7) & ~(size_t)7;
    float* p = ws + off; off += n; return p;
  };
  float* bufFS = alloc((size_t)N * 512);       // src-projected feats (max HC=512)
  float* bufFD = alloc((size_t)N * 512);       // dst-projected feats
  float* bufX  = alloc((size_t)N * 512);       // attention output / LN buffer
  float* bufT  = alloc((size_t)N * 8 * 128);   // FF hidden (max 163840 x 128)
  float* bufH  = alloc((size_t)N * 64);        // node embedding h
  float* bufE  = alloc((size_t)Eed * 8);       // per-edge logits / exp
  float* bufM  = alloc((size_t)N * 8);         // segment max
  float* bufD  = alloc((size_t)N * 8);         // segment sum (denominator)
  float* bufG  = alloc((size_t)N);             // gate logits / exp
  float* bufGM = alloc(32);
  float* bufGD = alloc(32);
  float* bufP  = alloc(32 * 512);

  // packed f16 weights (WMMA B-fragment order)
  size_t poff = 0;
  _Float16* packBase = (_Float16*)alloc(70000); // 140000 halfs > total 122880
  auto palloc = [&](size_t nh) { _Float16* p = packBase + poff; poff += nh; return p; };
  _Float16* pA0s = palloc(64 * 128);
  _Float16* pA0d = palloc(64 * 128);
  _Float16* pF0W1 = palloc(64 * 128);
  _Float16* pF0W2 = palloc(128 * 64);
  _Float16* pF0W3 = palloc(128 * 64);
  _Float16* pSWs = palloc(64 * 512);
  _Float16* pSWd = palloc(64 * 512);
  _Float16* pFsW1 = palloc(64 * 128);
  _Float16* pFsW2 = palloc(128 * 64);

  auto pack = [&](const float* W, _Float16* Wp, int K, int Nc) {
    int th = K * Nc / 16;
    k_packB<<<(th + 255) / 256, 256, 0, stream>>>(W, Wp, K, Nc);
  };
  pack(a0Ws, pA0s, 64, 128);
  pack(a0Wd, pA0d, 64, 128);
  pack(f0W1, pF0W1, 64, 128);
  pack(f0W2, pF0W2, 128, 64);
  pack(f0W3, pF0W3, 128, 64);
  pack(sWs,  pSWs, 64, 512);
  pack(sWd,  pSWd, 64, 512);
  pack(fsW1, pFsW1, 64, 128);
  pack(fsW2, pFsW2, 128, 64);

  const int gridE = (Eed + 7) / 8;
  const float NEG = -3.4e38f;

  auto gatv2 = [&](const float* fsL, const float* fdL, const float* aL,
                   float* outX, int Hd) {
    int HC = Hd * 64;
    k_fill<<<(N * Hd + 255) / 256, 256, 0, stream>>>(bufM, NEG, N * Hd);
    k_fill<<<(N * Hd + 255) / 256, 256, 0, stream>>>(bufD, 0.f, N * Hd);
    k_fill<<<(N * HC + 255) / 256, 256, 0, stream>>>(outX, 0.f, N * HC);
    k_edge_logits<<<gridE, 256, 0, stream>>>(fsL, fdL, aL, src, dst, bufE, bufM, Eed, Hd);
    k_edge_exp<<<(Eed * Hd + 255) / 256, 256, 0, stream>>>(bufE, bufM, bufD, dst, Eed, Hd);
    k_edge_agg<<<gridE, 256, 0, stream>>>(fsL, bufE, src, dst, outX, Eed, Hd);
    k_scale_rows<<<(N * HC + 255) / 256, 256, 0, stream>>>(outX, bufD, N * HC, Hd);
  };
  auto gemm = [&](const float* A, const _Float16* Bp, const float* bias, float* C,
                  int M, int K, int Nc, int act) {
    dim3 grid(M / 16, Nc / 64);
    k_gemm_bias<<<grid, 32, 0, stream>>>(A, Bp, bias, C, M, K, Nc, act);
  };

  // ---- stage 1: up GATv2 (heads=1), K=10 projection done scalar ----
  k_up<<<(N * 64 + 255) / 256, 256, 0, stream>>>(feats, upWs, upbs, upWd, upbd,
                                                 bufFS, bufFD, N * 64);
  gatv2(bufFS, bufFD, upa, bufX, 1);                       // bufX = h (N,64)

  // ---- stage 2: attn0 (heads=2) + ff0 ----
  gemm(bufX, pA0s, a0bs, bufFS, N, 64, 128, 0);            // fs2
  gemm(bufX, pA0d, a0bd, bufFD, N, 64, 128, 0);            // fd2
  gatv2(bufFS, bufFD, a0a, bufX, 2);                       // bufX = (N,2,64)
  k_layernorm<<<(N * 2 + 7) / 8, 256, 0, stream>>>(bufX, f0g, f0b, N * 2);
  gemm(bufX, pF0W1, f0b1, bufT, N * 2, 64, 128, 1);        // gelu(x@W1+b1)
  gemm(bufT, pF0W2, f0b2, bufFS, N * 2, 128, 64, 0);       // (N,2,64) == (N,128)
  gemm(bufFS, pF0W3, f0b3, bufH, N, 128, 64, 0);           // h (N,64)

  // ---- stage 3: attn_seq (heads=8) + ff_seq ----
  gemm(bufH, pSWs, sbs, bufFS, N, 64, 512, 0);             // fs3
  gemm(bufH, pSWd, sbd, bufFD, N, 64, 512, 0);             // fd3
  gatv2(bufFS, bufFD, sa, bufX, 8);                        // bufX = (N,8,64)
  k_layernorm<<<(N * 8 + 7) / 8, 256, 0, stream>>>(bufX, fsg, fsb, N * 8);
  gemm(bufX, pFsW1, fsb1, bufT, N * 8, 64, 128, 1);        // gelu
  gemm(bufT, pFsW2, fsb2, bufFS, N * 8, 128, 64, 0);       // x4 (N,512)

  // ---- stage 4: global attention pooling + tanh ----
  k_fill<<<1, 64, 0, stream>>>(bufGM, NEG, 32);
  k_fill<<<1, 64, 0, stream>>>(bufGD, 0.f, 32);
  k_fill<<<(16384 + 255) / 256, 256, 0, stream>>>(bufP, 0.f, 16384);
  k_gate<<<(N + 7) / 8, 256, 0, stream>>>(bufFS, Wg, bg, gid, bufG, bufGM, N);
  k_gate_exp<<<(N + 255) / 256, 256, 0, stream>>>(bufG, bufGM, bufGD, gid, N);
  k_pool<<<(N * 512 + 255) / 256, 256, 0, stream>>>(bufFS, bufG, bufGD, gid, bufP, N);
  k_tanh_out<<<(out_size + 255) / 256, 256, 0, stream>>>(bufP, (float*)d_out, out_size);
}
